// AttentionNet_69114613728868
// MI455X (gfx1250) — matrix operand order repaired and checked by reading, not compiled, not run on hardware
//
#include <hip/hip_runtime.h>
#include <hip/hip_bf16.h>
#include <math.h>
#include <stdint.h>

typedef __attribute__((ext_vector_type(16))) _Float16 v16h;
typedef __attribute__((ext_vector_type(8)))  float    v8f;

#define B_  8
#define D_  256
#define N_  4096
#define T_  1024
#define SCALE 0.0625f   // 1/sqrt(256)

#define KSTR 132   // LDS row stride (floats) for 128-wide tiles: 8*132 % 64 == 32
#define QSTR 66    // LDS row stride (floats) for  64-wide tiles: 16*66 % 64 == 32

// ---- CDNA5 async global->LDS staging (guarded; falls back to sync copies) ----
#if defined(__has_builtin)
#  if __has_builtin(__builtin_amdgcn_global_load_async_to_lds_b128) && \
      __has_builtin(__builtin_amdgcn_global_load_async_to_lds_b64)
#    define ASYNC_LDS 1
#  endif
#  if __has_builtin(__builtin_amdgcn_s_wait_asynccnt)
#    define HAVE_WAIT_ASYNC 1
#  endif
#endif

typedef int a_v4i __attribute__((vector_size(16)));
typedef int a_v2i __attribute__((vector_size(8)));
typedef __attribute__((address_space(1))) a_v4i* g4p_t;
typedef __attribute__((address_space(3))) a_v4i* l4p_t;
typedef __attribute__((address_space(1))) a_v2i* g2p_t;
typedef __attribute__((address_space(3))) a_v2i* l2p_t;

__device__ __forceinline__ void cp16(const float* g, float* l) {
#ifdef ASYNC_LDS
  __builtin_amdgcn_global_load_async_to_lds_b128(
      (g4p_t)(a_v4i*)g, (l4p_t)(a_v4i*)l, 0, 0);
#else
  *(float4*)l = *(const float4*)g;
#endif
}
__device__ __forceinline__ void cp8(const float* g, float* l) {
#ifdef ASYNC_LDS
  __builtin_amdgcn_global_load_async_to_lds_b64(
      (g2p_t)(a_v2i*)g, (l2p_t)(a_v2i*)l, 0, 0);
#else
  *(float2*)l = *(const float2*)g;
#endif
}
// Wait until at most N of this wave's async copies remain outstanding.
template <int N>
__device__ __forceinline__ void async_wait() {
#ifdef ASYNC_LDS
#  ifdef HAVE_WAIT_ASYNC
  __builtin_amdgcn_s_wait_asynccnt(N);
#  else
  asm volatile("s_wait_asynccnt %0" :: "i"(N) : "memory");
#  endif
#endif
}

// ---------------------------------------------------------------------------
// Kernel 1: S[b][n][t] = scale * sum_d K[b][d][n] * Q[b][d][t]
// Block = 8 waves (4n x 2t), tile 128(n) x 64(t); k-step 32.
// Double-buffered async staging of fp32 K/Q tiles; f16 WMMA, f32 acc.
// ---------------------------------------------------------------------------
__global__ __launch_bounds__(256) void scores_kernel(const float* __restrict__ Kp,
                                                     const float* __restrict__ Qp,
                                                     float* __restrict__ Sp) {
  __shared__ float ldsK[2][32 * KSTR];
  __shared__ float ldsQ[2][32 * QSTR];

  const int tid  = threadIdx.x;
  const int lane = tid & 31;
  const int wave = tid >> 5;
  const int half = lane >> 4;
  const int m16  = lane & 15;
  const int wn = wave & 3, wt = wave >> 2;
  const int nb = blockIdx.x * 128;
  const int tb = blockIdx.y * 64;
  const int b  = blockIdx.z;

  const float* Kb = Kp + (size_t)b * D_ * N_;
  const float* Qb = Qp + (size_t)b * D_ * T_;
  float* Sb = Sp + (size_t)b * N_ * T_;

  // 8 async ops per thread per tile (K: 4x b128, Q: 4x b64)
  auto stage = [&](int kk, int buf) {
#pragma unroll
    for (int u = 0; u < 4; ++u) {
      const int ch = tid + 256 * u;
      const int r = ch >> 5, cc = (ch & 31) << 2;
      cp16(Kb + (size_t)(kk + r) * N_ + nb + cc, &ldsK[buf][r * KSTR + cc]);
    }
#pragma unroll
    for (int u = 0; u < 4; ++u) {
      const int ch = tid + 256 * u;
      const int r = ch >> 5, cc = (ch & 31) << 1;
      cp8(Qb + (size_t)(kk + r) * T_ + tb + cc, &ldsQ[buf][r * QSTR + cc]);
    }
  };

  stage(0, 0);
  v8f c[2][2] = {};
  const int steps = D_ / 32;
  for (int it = 0; it < steps; ++it) {
    const int buf = it & 1;
    if (it + 1 < steps) { stage(32 * (it + 1), buf ^ 1); async_wait<8>(); }
    else                { async_wait<0>(); }
    __syncthreads();   // tile `it` visible to all waves

    v16h a[2], bb[2];
    // A frag: lanes 0-15: K = {0..7,16..23}; lanes 16-31: K = {8..15,24..31}
#pragma unroll
    for (int i = 0; i < 2; ++i) {
      const int col = wn * 32 + 16 * i + m16;
#pragma unroll
      for (int j = 0; j < 16; ++j) {
        const int k = 8 * half + j + ((j < 8) ? 0 : 8);
        a[i][j] = (_Float16)ldsK[buf][k * KSTR + col];
      }
    }
    // B frag: lanes 0-15: K = {0..7,16..23}; lanes 16-31: K = {8..15,24..31}
#pragma unroll
    for (int i = 0; i < 2; ++i) {
      const int col = wt * 32 + 16 * i + m16;
#pragma unroll
      for (int j = 0; j < 16; ++j)
        bb[i][j] = (_Float16)ldsQ[buf][(8 * half + j + ((j < 8) ? 0 : 8)) * QSTR + col];
    }
#pragma unroll
    for (int i = 0; i < 2; ++i)
#pragma unroll
      for (int j = 0; j < 2; ++j)
        c[i][j] = __builtin_amdgcn_wmma_f32_16x16x32_f16(
            false, a[i], false, bb[j], (short)0, c[i][j], false, false);
    __syncthreads();   // all waves done reading buf before it is overwritten
  }
  // C/D: VGPR r -> row M = r + 8*half, col N = lane%16
#pragma unroll
  for (int i = 0; i < 2; ++i)
#pragma unroll
    for (int j = 0; j < 2; ++j)
#pragma unroll
      for (int r = 0; r < 8; ++r) {
        const int row = nb + wn * 32 + 16 * i + 8 * half + r;
        const int col = tb + wt * 32 + 16 * j + m16;
        Sb[(size_t)row * T_ + col] = c[i][j][r] * SCALE;
      }
}

// ---------------------------------------------------------------------------
// Kernel 2: softmax over N (in place) + argmax. Block = 64 t-cols x 4 n-parts.
// ---------------------------------------------------------------------------
__global__ __launch_bounds__(256) void softmax_kernel(float* __restrict__ Sp,
                                                      int* __restrict__ amax) {
  const int tx = threadIdx.x & 63;
  const int ty = threadIdx.x >> 6;
  const int t  = blockIdx.x * 64 + tx;
  const int b  = blockIdx.y;
  float* Sb = Sp + (size_t)b * N_ * T_;

  float m = -INFINITY, s = 0.0f;
  int idx = 0;
  for (int n = ty; n < N_; n += 4) {
    const float x = Sb[(size_t)n * T_ + t];
    if (x > m) { s *= __expf(m - x); m = x; idx = n; }
    s += __expf(x - m);
  }
  __shared__ float sm[4][64];
  __shared__ float ss[4][64];
  __shared__ int   si[4][64];
  sm[ty][tx] = m; ss[ty][tx] = s; si[ty][tx] = idx;
  __syncthreads();
  if (ty == 0) {
    float M = m, Sum = s; int I = idx;
#pragma unroll
    for (int k = 1; k < 4; ++k) {
      const float mk = sm[k][tx], sk = ss[k][tx];
      const int ik = si[k][tx];
      if (mk > M)       { Sum = Sum * __expf(M - mk) + sk; M = mk; I = ik; }
      else if (mk == M) { Sum += sk; if (ik < I) I = ik; }
      else              { Sum += sk * __expf(mk - M); }
    }
    sm[0][tx] = M; ss[0][tx] = Sum;
    amax[b * T_ + t] = I;
  }
  __syncthreads();
  const float M   = sm[0][tx];
  const float inv = 1.0f / ss[0][tx];
  for (int n = ty; n < N_; n += 4) {
    const size_t off = (size_t)n * T_ + t;
    Sb[off] = __expf(Sb[off] - M) * inv;
  }
}

// ---------------------------------------------------------------------------
// Kernel 3: R[b][dd][t] = sum_n V[b][dd][n] * A[b][n][t]  (K' = 4096)
// V rows are k-contiguous (wide global loads); A tile double-buffered in LDS.
// ---------------------------------------------------------------------------
__global__ __launch_bounds__(256) void out_kernel(const float* __restrict__ Vp,
                                                  const float* __restrict__ Ap,
                                                  float* __restrict__ Rp) {
  __shared__ float ldsA[2][32 * QSTR];

  const int tid  = threadIdx.x;
  const int lane = tid & 31;
  const int wave = tid >> 5;
  const int half = lane >> 4;
  const int m16  = lane & 15;
  const int wn = wave & 3, wt = wave >> 2;
  const int db = blockIdx.x * 128;
  const int tb = blockIdx.y * 64;
  const int b  = blockIdx.z;

  const float* Vb = Vp + (size_t)b * D_ * N_;
  const float* Ab = Ap + (size_t)b * N_ * T_;
  float* Rb = Rp + (size_t)b * (2 * D_) * T_;

  // 4 async ops per thread per tile (A: 4x b64)
  auto stage = [&](int kk, int buf) {
#pragma unroll
    for (int u = 0; u < 4; ++u) {
      const int ch = tid + 256 * u;
      const int r = ch >> 5, cc = (ch & 31) << 1;
      cp8(Ab + (size_t)(kk + r) * T_ + tb + cc, &ldsA[buf][r * QSTR + cc]);
    }
  };

  stage(0, 0);
  v8f c[2][2] = {};
  const int steps = N_ / 32;
  for (int it = 0; it < steps; ++it) {
    const int buf = it & 1;
    const int kk  = it * 32;
    if (it + 1 < steps) { stage(kk + 32, buf ^ 1); async_wait<4>(); }
    else                { async_wait<0>(); }
    __syncthreads();

    v16h a[2], bb[2];
    // V rows: k-contiguous runs of 8 -> wide global loads
#pragma unroll
    for (int i = 0; i < 2; ++i) {
      const float* src = Vb + (size_t)(db + wn * 32 + 16 * i + m16) * N_ + kk + 8 * half;
#pragma unroll
      for (int j = 0; j < 8; ++j)  a[i][j] = (_Float16)src[j];        // K = 8h + j
#pragma unroll
      for (int j = 8; j < 16; ++j) a[i][j] = (_Float16)src[j + 8];    // K = 16 + 8h + (j-8)
    }
#pragma unroll
    for (int i = 0; i < 2; ++i) {
      const int col = wt * 32 + 16 * i + m16;
#pragma unroll
      for (int j = 0; j < 16; ++j)
        bb[i][j] = (_Float16)ldsA[buf][(8 * half + j + ((j < 8) ? 0 : 8)) * QSTR + col];
    }
#pragma unroll
    for (int i = 0; i < 2; ++i)
#pragma unroll
      for (int j = 0; j < 2; ++j)
        c[i][j] = __builtin_amdgcn_wmma_f32_16x16x32_f16(
            false, a[i], false, bb[j], (short)0, c[i][j], false, false);
    __syncthreads();
  }
#pragma unroll
  for (int i = 0; i < 2; ++i)
#pragma unroll
    for (int j = 0; j < 2; ++j)
#pragma unroll
      for (int r = 0; r < 8; ++r) {
        const int row = db + wn * 32 + 16 * i + 8 * half + r;
        const int col = tb + wt * 32 + 16 * j + m16;
        Rb[(size_t)row * T_ + col] = c[i][j][r];
      }
}

// ---------------------------------------------------------------------------
// Kernel 4: R_[b][D+dd][t] = Q[b][dd][t]  (vectorized)
// ---------------------------------------------------------------------------
__global__ __launch_bounds__(256) void qcopy_kernel(const float4* __restrict__ Qp,
                                                    float* __restrict__ Rp) {
  const int i = blockIdx.x * 256 + threadIdx.x;
  const int per_b = D_ * T_ / 4;
  if (i < B_ * per_b) {
    const int b = i / per_b, r = i % per_b;
    ((float4*)(Rp + (size_t)b * (2 * D_ * T_) + D_ * T_))[r] = Qp[i];
  }
}

extern "C" void kernel_launch(void* const* d_in, const int* in_sizes, int n_in,
                              void* d_out, int out_size, void* d_ws, size_t ws_size,
                              hipStream_t stream) {
  const float* Kp = (const float*)d_in[0];
  const float* Vp = (const float*)d_in[1];
  const float* Qp = (const float*)d_in[2];

  float* out = (float*)d_out;
  const size_t R_elems = (size_t)B_ * 2 * D_ * T_;   // 4,194,304
  const size_t A_elems = (size_t)B_ * N_ * T_;       // 33,554,432
  float* Rp = out;
  float* Ap = out + R_elems;                         // scores, then A in place
  int*   Mp = (int*)(out + R_elems + A_elems);       // maxAtt (int32)

  scores_kernel <<<dim3(N_ / 128, T_ / 64, B_), 256, 0, stream>>>(Kp, Qp, Ap);
  softmax_kernel<<<dim3(T_ / 64, B_),           256, 0, stream>>>(Ap, Mp);
  out_kernel    <<<dim3(2 * D_ / 128, T_ / 64, B_), 256, 0, stream>>>(Vp, Ap, Rp);
  qcopy_kernel  <<<dim3((B_ * D_ * T_ / 4 + 255) / 256), 256, 0, stream>>>(
      (const float4*)Qp, Rp);
}
